// Hybrid_GCNH_56882546868342
// MI455X (gfx1250) — matrix operand order repaired and checked by reading, not compiled, not run on hardware
//
#include <hip/hip_runtime.h>
#include <math.h>

// ---------------------------------------------------------------------------
// Hybrid GCNH forward for MI455X (gfx1250, wave32).
//  * one-time f32->f16 conversion of all operands (halves L2/HBM + LDS traffic)
//  * all GEMMs via v_wmma_f32_16x16x32_f16, f32 accumulate
//  * A-panel staged with global_load_async_to_lds_b128 (ASYNCcnt) when the
//    toolchain exposes it, double-buffered against WMMA compute
// ---------------------------------------------------------------------------

typedef __attribute__((ext_vector_type(16))) _Float16 v16h;
typedef __attribute__((ext_vector_type(8)))  float    v8f;
typedef __attribute__((ext_vector_type(4)))  _Float16 v4h;
typedef __attribute__((ext_vector_type(4)))  int      v4i;

#define NROWS 4096
#define FDIM  1024
#define HDIM  512
#define NCLS  16

// Block tile 128x128, 8 waves (4x2), per-wave 32x64 = 2x4 WMMA tiles, K-step 32.
#define TM 128
#define TN 128
#define TK 32
#define LDA 40   // halves; 80B row keeps 16B chunks aligned + conflict-free frag reads
#define LDB 40

union Frag16 { v16h v; unsigned int u[8]; };

#define EP_NONE       0
#define EP_BIAS       1
#define EP_BIAS_RELU  2
#define EP_BLEND_RELU 3   // relu(sig(*logit)*blend_src + (1-sig)*acc)

#if defined(__gfx1250__) && __has_builtin(__builtin_amdgcn_global_load_async_to_lds_b128) && __has_builtin(__builtin_amdgcn_s_wait_asynccnt)
#define ASYNC_COPY 1
#else
#define ASYNC_COPY 0
#endif

#if ASYNC_COPY
typedef __attribute__((address_space(1))) void* gas_vptr;
typedef __attribute__((address_space(3))) void* las_vptr;
typedef __attribute__((address_space(1))) v4i*  gas_v4p;
typedef __attribute__((address_space(3))) v4i*  las_v4p;

__device__ __forceinline__ void issue_a_async(const _Float16* A, _Float16* lbase,
                                              int m_blk, int K, int k0,
                                              int wave, int lane)
{
#pragma unroll
    for (int i = 0; i < 2; ++i) {           // 2 x b128 per wave = full 128x32 tile
        const int g   = wave * 64 + i * 32 + lane;   // chunk id 0..511
        const int row = g >> 2;
        const int ch  = g & 3;                        // 16B chunk within 64B row
        const _Float16* gp = &A[(size_t)(m_blk + row) * K + k0 + ch * 8];
        _Float16* lp = &lbase[row * LDA + ch * 8];
        __builtin_amdgcn_global_load_async_to_lds_b128(
            (gas_v4p)(gas_vptr)gp, (las_v4p)(las_vptr)lp, 0, 0);
    }
}
#endif

__launch_bounds__(256, 2)
__global__ void gemm_wmma_f16(const _Float16* __restrict__ A, const _Float16* __restrict__ B,
                              _Float16* __restrict__ C, int M, int Nn, int K,
                              const float* __restrict__ bias,
                              const _Float16* __restrict__ blend_src,
                              const float* __restrict__ blend_logit, int ep)
{
    __shared__ _Float16 As[2][TM * LDA];   // [m][k]
    __shared__ _Float16 Bs[2][TN * LDB];   // transposed: [n][k]

    const int tid  = threadIdx.x;
    const int lane = tid & 31;
    const int wave = tid >> 5;
    const int wrow = wave >> 1;            // 0..3
    const int wcol = wave & 1;             // 0..1
    const int m_blk = blockIdx.y * TM;
    const int n_blk = blockIdx.x * TN;

    const int mloc  = lane & 15;
    const int half  = lane >> 4;
    const int koffA = half * 8;            // A frag: lanes16-31 K {8..15,24..31}
    const int koffB = half * 8;            // B frag: lanes16-31 K {8..15,24..31}

    // B staging: thread -> k-row pair rp(0..15), 8 consecutive columns
    const int rp   = tid >> 4;
    const int col8 = (tid & 15) * 8;
#if !ASYNC_COPY
    const int arow_s = tid >> 1;           // 0..127
    const int ach_s  = (tid & 1) * 2;      // chunk 0/2 (2 consecutive 16B chunks)
#endif

    v8f acc[2][4] = {};
    const int nK = K / TK;

    // ---- prolog: tile 0 ----
    uint4 br0 = *(const uint4*)&B[(size_t)(2 * rp    ) * Nn + n_blk + col8];
    uint4 br1 = *(const uint4*)&B[(size_t)(2 * rp + 1) * Nn + n_blk + col8];
#if ASYNC_COPY
    issue_a_async(A, As[0], m_blk, K, 0, wave, lane);
#else
    uint4 ar0 = *(const uint4*)&A[(size_t)(m_blk + arow_s) * K + ach_s * 8];
    uint4 ar1 = *(const uint4*)&A[(size_t)(m_blk + arow_s) * K + ach_s * 8 + 8];
#endif

    for (int kt = 0; kt < nK; ++kt) {
        const int cur = kt & 1, nxt = cur ^ 1;
#if ASYNC_COPY
        __builtin_amdgcn_s_wait_asynccnt(0);     // A[kt] landed in As[cur]
#endif
        __syncthreads();                         // (1) prev compute reads done
#if ASYNC_COPY
        if (kt + 1 < nK) issue_a_async(A, As[nxt], m_blk, K, (kt + 1) * TK, wave, lane);
#else
        *(uint4*)&As[cur][arow_s * LDA + ach_s * 8]     = ar0;
        *(uint4*)&As[cur][arow_s * LDA + ach_s * 8 + 8] = ar1;
#endif
        // store current B with packed transpose (2 k-rows -> u32 pairs)
        {
            const unsigned r0w[4] = {br0.x, br0.y, br0.z, br0.w};
            const unsigned r1w[4] = {br1.x, br1.y, br1.z, br1.w};
#pragma unroll
            for (int i = 0; i < 8; ++i) {
                const unsigned lo = (r0w[i >> 1] >> ((i & 1) * 16)) & 0xffffu;
                const unsigned hi = (r1w[i >> 1] >> ((i & 1) * 16));
                *(unsigned*)&Bs[cur][(col8 + i) * LDB + 2 * rp] = lo | (hi << 16);
            }
        }
        // prefetch next tile's B (and A on sync path)
        if (kt + 1 < nK) {
            const int k0 = (kt + 1) * TK;
            br0 = *(const uint4*)&B[(size_t)(k0 + 2 * rp    ) * Nn + n_blk + col8];
            br1 = *(const uint4*)&B[(size_t)(k0 + 2 * rp + 1) * Nn + n_blk + col8];
#if !ASYNC_COPY
            ar0 = *(const uint4*)&A[(size_t)(m_blk + arow_s) * K + k0 + ach_s * 8];
            ar1 = *(const uint4*)&A[(size_t)(m_blk + arow_s) * K + k0 + ach_s * 8 + 8];
#endif
        }
        __syncthreads();                         // (2) As[cur]/Bs[cur] visible

        Frag16 af[2], bf[4];
        const _Float16* Ab = &As[cur][0];
        const _Float16* Bb = &Bs[cur][0];
#pragma unroll
        for (int mt = 0; mt < 2; ++mt) {
            const int row = wrow * 32 + mt * 16 + mloc;
#pragma unroll
            for (int j = 0; j < 8; ++j) {
                const int k = (j < 4) ? (koffA + 2 * j) : (16 + koffA + 2 * (j - 4));
                af[mt].u[j] = *(const unsigned*)&Ab[row * LDA + k];
            }
        }
#pragma unroll
        for (int nt = 0; nt < 4; ++nt) {
            const int col = wcol * 64 + nt * 16 + mloc;
#pragma unroll
            for (int j = 0; j < 8; ++j)
                bf[nt].u[j] = *(const unsigned*)&Bb[col * LDB + ((j < 4) ? (koffB + 2 * j) : (16 + koffB + 2 * (j - 4)))];
        }
#pragma unroll
        for (int mt = 0; mt < 2; ++mt)
#pragma unroll
            for (int nt = 0; nt < 4; ++nt)
                acc[mt][nt] = __builtin_amdgcn_wmma_f32_16x16x32_f16(
                    false, af[mt].v, false, bf[nt].v, (short)0, acc[mt][nt], false, false);
    }

    float bmix = 0.f;
    if (ep == EP_BLEND_RELU) bmix = 1.f / (1.f + __expf(-blend_logit[0]));

#pragma unroll
    for (int mt = 0; mt < 2; ++mt) {
#pragma unroll
        for (int nt = 0; nt < 4; ++nt) {
            const int c = n_blk + wcol * 64 + nt * 16 + mloc;
            const float bs = (ep == EP_BIAS || ep == EP_BIAS_RELU) ? bias[c] : 0.f;
            v8f t = acc[mt][nt];
            const float* tp = (const float*)&t;
#pragma unroll
            for (int v = 0; v < 8; ++v) {
                const int r = m_blk + wrow * 32 + mt * 16 + half * 8 + v;
                float x = tp[v] + bs;
                if (ep == EP_BLEND_RELU)
                    x = bmix * (float)blend_src[(size_t)r * Nn + c] + (1.f - bmix) * x;
                if (ep == EP_BIAS_RELU || ep == EP_BLEND_RELU) x = fmaxf(x, 0.f);
                C[(size_t)r * Nn + c] = (_Float16)x;
            }
        }
    }
}

// one-time f32 -> f16 conversion (vectorized)
__global__ void cvt_f32_f16(const float* __restrict__ s, _Float16* __restrict__ d, int n)
{
    const int i = (blockIdx.x * 256 + threadIdx.x) * 4;
    if (i < n) {
        const float4 v = *(const float4*)&s[i];
        v4h h; h.x = (_Float16)v.x; h.y = (_Float16)v.y; h.z = (_Float16)v.z; h.w = (_Float16)v.w;
        *(v4h*)&d[i] = h;
    }
}

// att[row] = sum_n q[n] * tanh(T[row,n] + bw[n] + b_att[n]); one wave per row
__global__ void att_kernel(const _Float16* __restrict__ T, const float* __restrict__ bw,
                           const float* __restrict__ b_att, const float* __restrict__ q,
                           float* __restrict__ att, int view)
{
    const int lane = threadIdx.x & 31;
    const int wave = threadIdx.x >> 5;
    const int row  = blockIdx.x * 8 + wave;
    float s = 0.f;
    for (int n = lane; n < HDIM; n += 32)
        s += q[n] * tanhf((float)T[(size_t)row * HDIM + n] + bw[n] + b_att[n]);
#pragma unroll
    for (int off = 16; off > 0; off >>= 1) s += __shfl_xor(s, off, 32);
    if (lane == 0) att[row * 3 + view] = s;
}

// fused = softmax3(att) . [hp, hp_knn, hp_ppr]
__global__ void fuse_kernel(const _Float16* __restrict__ hp, const _Float16* __restrict__ hk,
                            const _Float16* __restrict__ hpp, const float* __restrict__ att,
                            _Float16* __restrict__ fused)
{
    const size_t idx = (size_t)blockIdx.x * 256 + threadIdx.x;   // over NROWS*HDIM
    const int m = (int)(idx >> 9);
    const float a0 = att[m * 3], a1 = att[m * 3 + 1], a2 = att[m * 3 + 2];
    const float mx = fmaxf(a0, fmaxf(a1, a2));
    const float e0 = __expf(a0 - mx), e1 = __expf(a1 - mx), e2 = __expf(a2 - mx);
    const float inv = 1.f / (e0 + e1 + e2);
    fused[idx] = (_Float16)((e0 * (float)hp[idx] + e1 * (float)hk[idx] + e2 * (float)hpp[idx]) * inv);
}

// out = log_softmax(fused @ Wc + bc); 128 threads = 8 rows x 16 classes
__global__ void cls_kernel(const _Float16* __restrict__ fused, const float* __restrict__ Wc,
                           const float* __restrict__ bc, float* __restrict__ out)
{
    __shared__ float red[8][NCLS];
    const int c  = threadIdx.x & 15;
    const int rl = threadIdx.x >> 4;
    const int row = blockIdx.x * 8 + rl;
    float acc = bc[c];
    for (int k = 0; k < HDIM; ++k)
        acc += (float)fused[(size_t)row * HDIM + k] * Wc[k * NCLS + c];
    red[rl][c] = acc;
    __syncthreads();
    float mx = -1e30f;
#pragma unroll
    for (int i = 0; i < NCLS; ++i) mx = fmaxf(mx, red[rl][i]);
    float s = 0.f;
#pragma unroll
    for (int i = 0; i < NCLS; ++i) s += __expf(red[rl][i] - mx);
    out[(size_t)row * NCLS + c] = acc - mx - logf(s);
}

static inline void cvt(const float* s, _Float16* d, int n, hipStream_t stream)
{
    cvt_f32_f16<<<(n / 4 + 255) / 256, 256, 0, stream>>>(s, d, n);
}

static inline void launch_gemm(const _Float16* A, const _Float16* B, _Float16* C,
                               int M, int Nn, int K,
                               const float* bias, const _Float16* bsrc, const float* blog,
                               int ep, hipStream_t stream)
{
    dim3 grid(Nn / TN, M / TM), blk(256);
    gemm_wmma_f16<<<grid, blk, 0, stream>>>(A, B, C, M, Nn, K, bias, bsrc, blog, ep);
}

extern "C" void kernel_launch(void* const* d_in, const int* in_sizes, int n_in,
                              void* d_out, int out_size, void* d_ws, size_t ws_size,
                              hipStream_t stream)
{
    (void)in_sizes; (void)n_in; (void)out_size; (void)ws_size;
    const float* feat       = (const float*)d_in[0];
    const float* adj        = (const float*)d_in[1];
    const float* knn_adj    = (const float*)d_in[2];
    const float* ppr_adj    = (const float*)d_in[3];
    const float* fused_feat = (const float*)d_in[4];
    const float* Ws0 = (const float*)d_in[5];
    const float* Wn0 = (const float*)d_in[6];
    const float* Ws1 = (const float*)d_in[7];
    const float* Wn1 = (const float*)d_in[8];
    const float* bl0 = (const float*)d_in[9];
    const float* bl1 = (const float*)d_in[10];
    const float* Wg  = (const float*)d_in[11];
    const float* bg  = (const float*)d_in[12];
    const float* Wa  = (const float*)d_in[13];
    const float* ba  = (const float*)d_in[14];
    const float* Ww  = (const float*)d_in[15];
    const float* bw  = (const float*)d_in[16];
    const float* b_att = (const float*)d_in[17];
    const float* q   = (const float*)d_in[18];
    const float* Wc  = (const float*)d_in[19];
    const float* bc  = (const float*)d_in[20];

    // Workspace layout (f16 unless noted), ~112.1 MB peak, all written before read
    char* ws = (char*)d_ws;
    #define WSOFF(mb) ((size_t)(mb) << 20)
    _Float16* ADJ_H   = (_Float16*)(ws + WSOFF(0));    // 32 MB
    _Float16* KP_H    = (_Float16*)(ws + WSOFF(32));   // 32 MB (knn, then ppr)
    _Float16* FEAT_H  = (_Float16*)(ws + WSOFF(64));   //  8 MB
    _Float16* FUSED_H = (_Float16*)(ws + WSOFF(72));   //  8 MB
    _Float16* WS0H = (_Float16*)(ws + WSOFF(80));
    _Float16* WN0H = (_Float16*)(ws + WSOFF(81));
    _Float16* WS1H = (_Float16*)(ws + WSOFF(82));
    _Float16* WN1H = (_Float16*)(ws + WSOFF(83));
    _Float16* WGH  = (_Float16*)(ws + WSOFF(84));
    _Float16* WAH  = (_Float16*)(ws + WSOFF(85));
    _Float16* WWH  = (_Float16*)(ws + WSOFF(86));
    _Float16* S0 = (_Float16*)(ws + WSOFF(88));        // 8 MB: AX/AH0/G/Aa/T
    _Float16* S1 = (_Float16*)(ws + WSOFF(96));        // 4 MB: XS0/HS1/fused
    _Float16* S2 = (_Float16*)(ws + WSOFF(100));       // 4 MB: h0 then hp
    _Float16* S3 = (_Float16*)(ws + WSOFF(104));       // 4 MB: hp_knn
    _Float16* S4 = (_Float16*)(ws + WSOFF(108));       // 4 MB: hp_ppr
    float*    ATT = (float*)(ws + WSOFF(112));         // 48 KB

    // one-time conversions
    cvt(feat, FEAT_H, NROWS * FDIM, stream);
    cvt(fused_feat, FUSED_H, NROWS * FDIM, stream);
    cvt(Ws0, WS0H, FDIM * HDIM, stream);
    cvt(Wn0, WN0H, FDIM * HDIM, stream);
    cvt(Ws1, WS1H, HDIM * HDIM, stream);
    cvt(Wn1, WN1H, HDIM * HDIM, stream);
    cvt(Wg,  WGH,  FDIM * HDIM, stream);
    cvt(Wa,  WAH,  FDIM * HDIM, stream);
    cvt(Ww,  WWH,  HDIM * HDIM, stream);
    cvt(adj, ADJ_H, NROWS * NROWS, stream);

    // GCNH layer 0: h0 = relu(sig(bl0)*feat@Ws0 + (1-sig)*((adj@feat)@Wn0))
    launch_gemm(FEAT_H, WS0H, S1, NROWS, HDIM, FDIM, 0, 0, 0, EP_NONE, stream);        // XS0
    launch_gemm(ADJ_H, FEAT_H, S0, NROWS, FDIM, NROWS, 0, 0, 0, EP_NONE, stream);      // AX
    launch_gemm(S0, WN0H, S2, NROWS, HDIM, FDIM, 0, S1, bl0, EP_BLEND_RELU, stream);   // h0
    // GCNH layer 1
    launch_gemm(S2, WS1H, S1, NROWS, HDIM, HDIM, 0, 0, 0, EP_NONE, stream);            // HS1
    launch_gemm(ADJ_H, S2, S0, NROWS, HDIM, NROWS, 0, 0, 0, EP_NONE, stream);          // AH0
    launch_gemm(S0, WN1H, S2, NROWS, HDIM, HDIM, 0, S1, bl1, EP_BLEND_RELU, stream);   // hp
    // KNN view
    launch_gemm(FUSED_H, WGH, S0, NROWS, HDIM, FDIM, 0, 0, 0, EP_NONE, stream);        // G
    cvt(knn_adj, KP_H, NROWS * NROWS, stream);
    launch_gemm(KP_H, S0, S3, NROWS, HDIM, NROWS, bg, 0, 0, EP_BIAS_RELU, stream);     // hp_knn
    // PPR view
    launch_gemm(FEAT_H, WAH, S0, NROWS, HDIM, FDIM, ba, 0, 0, EP_BIAS, stream);        // Aa
    cvt(ppr_adj, KP_H, NROWS * NROWS, stream);
    launch_gemm(KP_H, S0, S4, NROWS, HDIM, NROWS, 0, 0, 0, EP_NONE, stream);           // hp_ppr
    // attention scores
    launch_gemm(S2, WWH, S0, NROWS, HDIM, HDIM, 0, 0, 0, EP_NONE, stream);
    att_kernel<<<NROWS / 8, 256, 0, stream>>>(S0, bw, b_att, q, ATT, 0);
    launch_gemm(S3, WWH, S0, NROWS, HDIM, HDIM, 0, 0, 0, EP_NONE, stream);
    att_kernel<<<NROWS / 8, 256, 0, stream>>>(S0, bw, b_att, q, ATT, 1);
    launch_gemm(S4, WWH, S0, NROWS, HDIM, HDIM, 0, 0, 0, EP_NONE, stream);
    att_kernel<<<NROWS / 8, 256, 0, stream>>>(S0, bw, b_att, q, ATT, 2);
    // fuse + classifier
    fuse_kernel<<<(NROWS * HDIM) / 256, 256, 0, stream>>>(S2, S3, S4, ATT, S1);
    cls_kernel<<<NROWS / 8, 128, 0, stream>>>(S1, Wc, bc, (float*)d_out);
}
